// HModel_31748398252728
// MI455X (gfx1250) — matrix operand order repaired and checked by reading, not compiled, not run on hardware
//
#include <hip/hip_runtime.h>

typedef __attribute__((ext_vector_type(16))) _Float16 v16h;
typedef __attribute__((ext_vector_type(8)))  _Float16 v8h;
typedef __attribute__((ext_vector_type(8)))  float    v8f;

// Problem constants (fixed by the reference harness)
constexpr int    cNH   = 100000;        // N_H source nodes
constexpr int    cE    = 3200000;       // edges
constexpr int    cDPAD = 224;           // 201 padded to 7*32
constexpr size_t cMM   = (size_t)cNH * 32;   // one moment plane (floats)

// d_ws layout (bytes)
constexpr size_t OFF_WF   = 0;                               // 40 KB of f16 weight frags
constexpr size_t OFF_CNT  = 64 * 1024;                       // cnt[N_H] f32
constexpr size_t OFF_S    = OFF_CNT + (size_t)cNH * 4;       // S[4][N_H*32] f32
constexpr size_t OFF_FEAT = OFF_S + 4 * cMM * 4;             // feat[N_H][224] f16

// Weight fragment offsets (in halfs); each frag = 32 lanes * 16 halfs = 512 halfs
constexpr int FR_W1A = 0;        // 2 frags
constexpr int FR_W1B = 1024;     // 2 frags
constexpr int FR_W2A = 2048;     // 28 frags (7 kc x 4 nc)
constexpr int FR_W2B = 16384;    // 8 frags (2 kc x 4 nc)

// leaky(x) with slope 0.1 < 1 is exactly max(x, 0.1x): lowers to v_mul + v_max_num_f32
__device__ __forceinline__ float leaky(float x) { return fmaxf(x, 0.1f * x); }

__device__ __forceinline__ v16h load_fragB(const _Float16* base, int lane) {
    // fragment-linear: lane-major, 16 halfs (32 B) per lane -> 2x global_load_b128
    return *(const v16h*)(base + (size_t)lane * 16);
}

__device__ __forceinline__ v8f wmma_f16(v16h a, v16h b, v8f c) {
    return __builtin_amdgcn_wmma_f32_16x16x32_f16(false, a, false, b, (short)0, c, false, false);
}

__device__ __forceinline__ void atomAddF(float* p, float v) {
    unsafeAtomicAdd(p, v);   // global_atomic_add_f32, non-returning
}

// ---------------------------------------------------------------------------
// Kernel 1: convert weights f32 -> f16 in WMMA B-fragment-linear layout.
// B frag layout: lane n(0..15): N=n, K = kc*32 + {0..7,16..23} ; lane n+16: K = kc*32 + {8..15,24..31}
// ---------------------------------------------------------------------------
__global__ void prep_weights(const float* __restrict__ W1a, const float* __restrict__ W1b,
                             const float* __restrict__ W2a, const float* __restrict__ W2b,
                             _Float16* __restrict__ out) {
    int t = blockIdx.x * blockDim.x + threadIdx.x;     // 0 .. 40*512-1
    if (t >= 40 * 512) return;
    int frag = t >> 9;
    int idx  = t & 511;
    int lane = idx >> 4, h = idx & 15;
    const float* W; int Krows, Ncols, kc, nc;
    if (frag < 2)       { W = W1a; Krows = 32;  Ncols = 32; kc = 0;              nc = frag; }
    else if (frag < 4)  { W = W1b; Krows = 32;  Ncols = 32; kc = 0;              nc = frag - 2; }
    else if (frag < 32) { int f = frag - 4;  W = W2a; Krows = 201; Ncols = 64; kc = f >> 2; nc = f & 3; }
    else                { int f = frag - 32; W = W2b; Krows = 64;  Ncols = 64; kc = f >> 2; nc = f & 3; }
    int K = kc * 32 + ((h >> 3) << 4) + ((lane >> 4) << 3) + (h & 7);
    int N = nc * 16 + (lane & 15);
    float v = (K < Krows) ? W[(size_t)K * Ncols + N] : 0.0f;
    out[t] = (_Float16)v;
}

// ---------------------------------------------------------------------------
// Kernel 2: edge MLP + one-pass raw-moment scatter (atomics).
// One wave processes one 16-edge tile per loop step.
// ---------------------------------------------------------------------------
__global__ void edge_kernel(const float* __restrict__ x_g, const float* __restrict__ eattr,
                            const int* __restrict__ eidx,   // [2][E]: src row, tgt row
                            const float* __restrict__ b1a, const float* __restrict__ b1b,
                            const _Float16* __restrict__ wf,
                            float* __restrict__ cnt, float* __restrict__ S) {
    __shared__ __align__(16) _Float16 ltile[4][16 * 32];   // per-wave hidden tile (1 KB)
    __shared__ __align__(16) int      lsrc[4][16];

    const int tid  = threadIdx.x;
    const int lane = tid & 31;
    const int wib  = tid >> 5;
    const int l16  = lane & 15;
    const int hi   = lane >> 4;                            // 0 or 1
    const int gwave = (blockIdx.x * blockDim.x + tid) >> 5;
    const int nwave = (gridDim.x * blockDim.x) >> 5;

    // weights resident in VGPRs for the whole loop
    v16h B1a0 = load_fragB(wf + FR_W1A,       lane);
    v16h B1a1 = load_fragB(wf + FR_W1A + 512, lane);
    v16h B1b0 = load_fragB(wf + FR_W1B,       lane);
    v16h B1b1 = load_fragB(wf + FR_W1B + 512, lane);
    const float ba0 = b1a[l16], ba1 = b1a[16 + l16];
    const float bb0 = b1b[l16], bb1 = b1b[16 + l16];

    const int* srcp = eidx;
    const int* tgtp = eidx + cE;
    _Float16* myt = ltile[wib];
    int*      mys = lsrc[wib];
    const int nTiles = cE / 16;

    for (int t = gwave; t < nTiles; t += nwave) {
        const int e  = t * 16 + l16;
        const int tg = tgtp[e];
        const int mySrc = srcp[e];
        if (hi == 0) mys[l16] = mySrc;

        // Prefetch next grid-stride tile's streaming data (global_prefetch_b8)
        const int tn = t + nwave;
        if (tn < nTiles) {
            __builtin_prefetch(eattr + (size_t)(tn * 16 + l16) * 16 + hi * 8, 0, 1);
            __builtin_prefetch(tgtp + tn * 16 + l16, 0, 1);
            __builtin_prefetch(srcp + tn * 16 + l16, 0, 1);
        }

        // Build A fragment: lane row M=l16; hi selects K halves {0..7}/{8..15} of each 16-block
        const float4* xg = (const float4*)(x_g + (size_t)tg * 16);
        const float4* ea = (const float4*)(eattr + (size_t)e * 16);
        float4 g0 = xg[hi * 2], g1 = xg[hi * 2 + 1];
        float4 a0 = ea[hi * 2], a1 = ea[hi * 2 + 1];
        float fin[16] = { g0.x, g0.y, g0.z, g0.w, g1.x, g1.y, g1.z, g1.w,
                          a0.x, a0.y, a0.z, a0.w, a1.x, a1.y, a1.z, a1.w };
        v16h A;
        #pragma unroll
        for (int j = 0; j < 16; j++) A[j] = (_Float16)fin[j];

        v8f c0, c1;
        #pragma unroll
        for (int i = 0; i < 8; i++) { c0[i] = ba0; c1[i] = ba1; }
        c0 = wmma_f16(A, B1a0, c0);
        c1 = wmma_f16(A, B1a1, c1);

        // leaky + write hidden tile to LDS (D layout -> row-major f16)
        #pragma unroll
        for (int i = 0; i < 8; i++) {
            int M = i + hi * 8;
            myt[M * 32 + l16]      = (_Float16)leaky(c0[i]);
            myt[M * 32 + 16 + l16] = (_Float16)leaky(c1[i]);
        }
        asm volatile("s_wait_dscnt 0" ::: "memory");

        // Reload as A fragment for layer 2
        const v8h* row = (const v8h*)(myt + (size_t)l16 * 32);
        v8h q0 = row[hi], q1 = row[2 + hi];
        v16h A2;
        #pragma unroll
        for (int j = 0; j < 8; j++) { A2[j] = q0[j]; A2[8 + j] = q1[j]; }

        v8f d0, d1;
        #pragma unroll
        for (int i = 0; i < 8; i++) { d0[i] = bb0; d1[i] = bb1; }
        d0 = wmma_f16(A2, B1b0, d0);
        d1 = wmma_f16(A2, B1b1, d1);

        // Gather the 8 src ids this lane needs (edges M = hi*8 + 0..7)
        int4 sA = *(const int4*)(mys + hi * 8);
        int4 sB = *(const int4*)(mys + hi * 8 + 4);
        int sv[8] = { sA.x, sA.y, sA.z, sA.w, sB.x, sB.y, sB.z, sB.w };

        // Scatter raw moments 1..4 (dims N = l16 and 16+l16)
        // Within one atomic instruction, lanes 0-15 share an edge -> 64B-contiguous
        #pragma unroll
        for (int i = 0; i < 8; i++) {
            const size_t o0 = (size_t)sv[i] * 32 + l16;
            const size_t o1 = o0 + 16;
            float v0 = d0[i], v1 = d1[i];
            float v02 = v0 * v0, v12 = v1 * v1;
            atomAddF(S + o0,            v0);
            atomAddF(S + cMM + o0,      v02);
            atomAddF(S + 2 * cMM + o0,  v02 * v0);
            atomAddF(S + 3 * cMM + o0,  v02 * v02);
            atomAddF(S + o1,            v1);
            atomAddF(S + cMM + o1,      v12);
            atomAddF(S + 2 * cMM + o1,  v12 * v1);
            atomAddF(S + 3 * cMM + o1,  v12 * v12);
        }
        if (hi == 0) atomAddF(cnt + mySrc, 1.0f);
    }
}

// ---------------------------------------------------------------------------
// Kernel 3: per-node stats finalize + f16 feature assembly. One wave per node.
// feat = [x_h(64), cnt, a(32), b(32), c(32), d(32), u(8), pad(23)]
// ---------------------------------------------------------------------------
__global__ void finalize_kernel(const float* __restrict__ x_h, const float* __restrict__ u,
                                const int* __restrict__ batch_h,
                                const float* __restrict__ cnt, const float* __restrict__ S,
                                _Float16* __restrict__ feat) {
    const int lane = threadIdx.x & 31;
    const int node = blockIdx.x * (blockDim.x >> 5) + (threadIdx.x >> 5);
    if (node >= cNH) return;

    const float n = cnt[node];
    const float denom = fmaxf(n, 1.0f);
    const size_t o = (size_t)node * 32 + lane;
    const float s1 = S[o], s2 = S[cMM + o], s3 = S[2 * cMM + o], s4 = S[3 * cMM + o];
    const float a  = s1 / denom;
    const float a2 = a * a, a3 = a2 * a, a4 = a2 * a2;
    float var = s2 / denom - a2; if (var < 0.0f) var = 0.0f;
    const float b  = sqrtf(1e-6f + var);
    const float m3 = (s3 - 3.0f * a * s2 + 3.0f * a2 * s1 - a3 * n) / denom;
    const float m4 = (s4 - 4.0f * a * s3 + 6.0f * a2 * s2 - 4.0f * a3 * s1 + a4 * n) / denom;
    const float cc = m3 / (b * b * b);
    const float dd = m4 / (b * b * b * b);

    _Float16* f = feat + (size_t)node * cDPAD;
    f[lane]        = (_Float16)x_h[(size_t)node * 64 + lane];
    f[32 + lane]   = (_Float16)x_h[(size_t)node * 64 + 32 + lane];
    if (lane == 0) f[64] = (_Float16)n;
    f[65 + lane]   = (_Float16)a;
    f[97 + lane]   = (_Float16)b;
    f[129 + lane]  = (_Float16)cc;
    f[161 + lane]  = (_Float16)dd;
    if (lane < 8)  f[193 + lane] = (_Float16)u[(size_t)batch_h[node] * 8 + lane];
    if (lane < 23) f[201 + lane] = (_Float16)0.0f;
}

// ---------------------------------------------------------------------------
// Kernel 4: node MLP 201(224)->64->64 via WMMA, 16 nodes per wave-tile.
// ---------------------------------------------------------------------------
__global__ void node_kernel(const _Float16* __restrict__ feat, const _Float16* __restrict__ wf,
                            const float* __restrict__ b2a, const float* __restrict__ b2b,
                            float* __restrict__ out) {
    __shared__ __align__(16) _Float16 ltile[4][16 * 64];   // per-wave hidden tile (2 KB)
    const int tid  = threadIdx.x;
    const int lane = tid & 31;
    const int wib  = tid >> 5;
    const int l16  = lane & 15;
    const int hi   = lane >> 4;
    const int gwave = (blockIdx.x * blockDim.x + tid) >> 5;
    const int nwave = (gridDim.x * blockDim.x) >> 5;

    // W2b frags register-resident (8 x 8 VGPRs)
    v16h B2b[8];
    #pragma unroll
    for (int fI = 0; fI < 8; fI++) B2b[fI] = load_fragB(wf + FR_W2B + fI * 512, lane);
    float biasA[4], biasB[4];
    #pragma unroll
    for (int nc = 0; nc < 4; nc++) { biasA[nc] = b2a[nc * 16 + l16]; biasB[nc] = b2b[nc * 16 + l16]; }

    _Float16* myt = ltile[wib];
    const int nTiles = cNH / 16;

    for (int t = gwave; t < nTiles; t += nwave) {
        const int node = t * 16 + l16;
        const _Float16* fr = feat + (size_t)node * cDPAD;

        // 7 A fragments from the f16 feature row
        v16h A[7];
        #pragma unroll
        for (int kc = 0; kc < 7; kc++) {
            const v8h* p = (const v8h*)(fr + kc * 32);
            v8h q0 = p[hi], q1 = p[2 + hi];
            #pragma unroll
            for (int j = 0; j < 8; j++) { A[kc][j] = q0[j]; A[kc][8 + j] = q1[j]; }
        }

        // Layer 1: 28 WMMAs -> leaky -> LDS tile
        #pragma unroll
        for (int nc = 0; nc < 4; nc++) {
            v8f acc;
            #pragma unroll
            for (int i = 0; i < 8; i++) acc[i] = biasA[nc];
            #pragma unroll
            for (int kc = 0; kc < 7; kc++) {
                v16h B = load_fragB(wf + FR_W2A + (kc * 4 + nc) * 512, lane);
                acc = wmma_f16(A[kc], B, acc);
            }
            #pragma unroll
            for (int i = 0; i < 8; i++) {
                int M = i + hi * 8;
                myt[M * 64 + nc * 16 + l16] = (_Float16)leaky(acc[i]);
            }
        }
        asm volatile("s_wait_dscnt 0" ::: "memory");

        // Hidden A fragments (K=64 -> 2 chunks)
        v16h H[2];
        #pragma unroll
        for (int kc = 0; kc < 2; kc++) {
            const v8h* p = (const v8h*)(myt + (size_t)l16 * 64 + kc * 32);
            v8h q0 = p[hi], q1 = p[2 + hi];
            #pragma unroll
            for (int j = 0; j < 8; j++) { H[kc][j] = q0[j]; H[kc][8 + j] = q1[j]; }
        }

        // Layer 2: 8 WMMAs -> f32 output (lanes 0-15 / 16-31 each store 64B-contiguous)
        #pragma unroll
        for (int nc = 0; nc < 4; nc++) {
            v8f acc;
            #pragma unroll
            for (int i = 0; i < 8; i++) acc[i] = biasB[nc];
            acc = wmma_f16(H[0], B2b[nc],     acc);
            acc = wmma_f16(H[1], B2b[4 + nc], acc);
            #pragma unroll
            for (int i = 0; i < 8; i++) {
                int M = i + hi * 8;
                out[(size_t)(t * 16 + M) * 64 + nc * 16 + l16] = acc[i];
            }
        }
    }
}

// ---------------------------------------------------------------------------
extern "C" void kernel_launch(void* const* d_in, const int* in_sizes, int n_in,
                              void* d_out, int out_size, void* d_ws, size_t ws_size,
                              hipStream_t stream) {
    const float* x_h  = (const float*)d_in[0];
    const float* x_g  = (const float*)d_in[1];
    const float* eatt = (const float*)d_in[2];
    const float* u    = (const float*)d_in[3];
    const int*   eidx = (const int*)d_in[4];
    const int*   bh   = (const int*)d_in[5];
    const float* W1a  = (const float*)d_in[6];
    const float* b1a  = (const float*)d_in[7];
    const float* W1b  = (const float*)d_in[8];
    const float* b1b  = (const float*)d_in[9];
    const float* W2a  = (const float*)d_in[10];
    const float* b2a  = (const float*)d_in[11];
    const float* W2b  = (const float*)d_in[12];
    const float* b2b  = (const float*)d_in[13];

    char* ws = (char*)d_ws;
    _Float16* wf   = (_Float16*)(ws + OFF_WF);
    float*    cnt  = (float*)(ws + OFF_CNT);
    float*    S    = (float*)(ws + OFF_S);
    _Float16* feat = (_Float16*)(ws + OFF_FEAT);
    float*    outp = (float*)d_out;

    // zero the atomic accumulators (cnt + 4 moment planes) every call
    hipMemsetAsync(ws + OFF_CNT, 0, (size_t)cNH * 4 + 4 * cMM * 4, stream);

    prep_weights<<<80, 256, 0, stream>>>(W1a, W1b, W2a, W2b, wf);
    edge_kernel<<<6400, 128, 0, stream>>>(x_g, eatt, eidx, b1a, b1b, wf, cnt, S);
    finalize_kernel<<<(cNH + 7) / 8, 256, 0, stream>>>(x_h, u, bh, cnt, S, feat);
    node_kernel<<<1563, 128, 0, stream>>>(feat, wf, b2a, b2b, outp);

    (void)in_sizes; (void)n_in; (void)out_size; (void)ws_size;
}
